// HypersphereSelfAttention_90452011254067
// MI455X (gfx1250) — matrix operand order repaired and checked by reading, not compiled, not run on hardware
//
#include <hip/hip_runtime.h>
#include <hip/hip_bf16.h>

typedef __bf16 v16bf __attribute__((ext_vector_type(16)));
typedef __bf16 v8bf  __attribute__((ext_vector_type(8)));
typedef float  v8f   __attribute__((ext_vector_type(8)));

#define D_MODEL 1024
#define SEQ     2048
#define BATCH   2
#define NHEAD   16
#define HD      64
#define NEG_BIG (-3.38953139e38f)

static __device__ __forceinline__ __bf16 f2bf(float f) {
  unsigned u = __builtin_bit_cast(unsigned, f);
  unsigned r = u + 0x7FFFu + ((u >> 16) & 1u);
  unsigned short hsv = (unsigned short)(r >> 16);
  return __builtin_bit_cast(__bf16, hsv);
}
static __device__ __forceinline__ float bf2f(__bf16 b) {
  unsigned short hsv = __builtin_bit_cast(unsigned short, b);
  unsigned u = ((unsigned)hsv) << 16;
  return __builtin_bit_cast(float, u);
}
static __device__ __forceinline__ v16bf combine16(v8bf lo, v8bf hi) {
  v16bf r;
#pragma unroll
  for (int i = 0; i < 8; ++i) { r[i] = lo[i]; r[i + 8] = hi[i]; }
  return r;
}
// low 32 bits of a flat pointer to __shared__ = LDS byte offset (aperture rule)
static __device__ __forceinline__ unsigned lds_off(const void* p) {
  return (unsigned)(size_t)p;
}
// async DMA: 32 contiguous bytes global -> LDS (two b128 transfers)
static __device__ __forceinline__ void async_copy32(unsigned lds, const void* g) {
  asm volatile("global_load_async_to_lds_b128 %0, %1, off\n\t"
               "global_load_async_to_lds_b128 %0, %1, off offset:16"
               :: "v"(lds), "v"(g) : "memory");
}
static __device__ __forceinline__ void wait_async0() {
  asm volatile("s_wait_asynccnt 0x0" ::: "memory");
}

// ---------------------------------------------------------------- converts
__global__ void f32_to_bf16_kernel(const float* __restrict__ in,
                                   __bf16* __restrict__ out, int n) {
  int i = blockIdx.x * blockDim.x + threadIdx.x;
  if (i < n) out[i] = f2bf(in[i]);
}

// ------------------------------------------------- per-head L2 normalization
__global__ void headnorm_kernel(__bf16* __restrict__ q, int nvec) {
  int i = blockIdx.x * blockDim.x + threadIdx.x;
  if (i >= nvec) return;
  __bf16* p = q + (size_t)i * HD;
  float ss = 0.f;
#pragma unroll 8
  for (int j = 0; j < HD; ++j) { float v = bf2f(p[j]); ss += v * v; }
  float inv = 1.0f / (sqrtf(ss) + 1e-6f);
#pragma unroll 8
  for (int j = 0; j < HD; ++j) p[j] = f2bf(bf2f(p[j]) * inv);
}

// -------------------------------------------------------------- WMMA GEMM
// C[M,N] = A[M,K] * Bw[N,K]^T  (bf16 in, f32 accumulate)
// 256 threads = 8 waves, tile 128x128, K-step 32, waves 2(M) x 4(N)
// double-buffered LDS filled by async global->LDS DMA (ASYNCcnt pipelined)
template <bool OUT_BF16>
__global__ __launch_bounds__(256) void gemm_bf16_wmma(
    const __bf16* __restrict__ A, const __bf16* __restrict__ Bw,
    float* __restrict__ Cf, __bf16* __restrict__ Cb, int M, int N, int K) {
  __shared__ __bf16 As[2][128 * 32];
  __shared__ __bf16 Bs[2][128 * 32];

  const int tid  = threadIdx.x;
  const int lane = tid & 31, hf = lane >> 4, l16 = lane & 15;
  const int w = tid >> 5, wm = w >> 2, wn = w & 3;
  const int m0 = blockIdx.y * 128, n0 = blockIdx.x * 128;

  v8f acc[4][2];
#pragma unroll
  for (int mi = 0; mi < 4; ++mi)
#pragma unroll
    for (int ni = 0; ni < 2; ++ni) acc[mi][ni] = {};

  const int lrow = tid >> 1, lseg = tid & 1;
  const __bf16* gA = A  + (size_t)(m0 + lrow) * K + lseg * 16;
  const __bf16* gB = Bw + (size_t)(n0 + lrow) * K + lseg * 16;
  const unsigned la0 = lds_off(&As[0][lrow * 32 + lseg * 16]);
  const unsigned la1 = lds_off(&As[1][lrow * 32 + lseg * 16]);
  const unsigned lb0 = lds_off(&Bs[0][lrow * 32 + lseg * 16]);
  const unsigned lb1 = lds_off(&Bs[1][lrow * 32 + lseg * 16]);

  // prologue: tile 0 -> buffer 0
  async_copy32(la0, gA);
  async_copy32(lb0, gB);

  const int nk = K / 32;
  for (int it = 0; it < nk; ++it) {
    const int cur = it & 1;
    wait_async0();        // tile `cur` resident in LDS (this wave's DMA done)
    __syncthreads();      // ...and every other wave's too; prev compute done
    if (it + 1 < nk) {    // kick off next tile into the other buffer
      const int k1 = (it + 1) * 32;
      async_copy32(cur ? la0 : la1, gA + k1);
      async_copy32(cur ? lb0 : lb1, gB + k1);
    }

    const __bf16* Ab_ = As[cur];
    const __bf16* Bb_ = Bs[cur];
    v16bf afrag[4], bfrag[2];
#pragma unroll
    for (int mi = 0; mi < 4; ++mi) {
      const __bf16* p = Ab_ + (wm * 64 + mi * 16 + l16) * 32 + hf * 8;
      afrag[mi] = combine16(*(const v8bf*)p, *(const v8bf*)(p + 16));
    }
#pragma unroll
    for (int ni = 0; ni < 2; ++ni) {
      const __bf16* p = Bb_ + (wn * 32 + ni * 16 + l16) * 32 + hf * 8;
      bfrag[ni] = combine16(*(const v8bf*)p, *(const v8bf*)(p + 16));
    }
#pragma unroll
    for (int mi = 0; mi < 4; ++mi)
#pragma unroll
      for (int ni = 0; ni < 2; ++ni)
        acc[mi][ni] = __builtin_amdgcn_wmma_f32_16x16x32_bf16(
            false, afrag[mi], false, bfrag[ni], (short)0, acc[mi][ni], false, false);
  }
  __syncthreads();

#pragma unroll
  for (int mi = 0; mi < 4; ++mi)
#pragma unroll
    for (int ni = 0; ni < 2; ++ni)
#pragma unroll
      for (int r = 0; r < 8; ++r) {
        int gm = m0 + wm * 64 + mi * 16 + r + hf * 8;
        int gn = n0 + wn * 32 + ni * 16 + l16;
        float v = acc[mi][ni][r];
        if (OUT_BF16) Cb[(size_t)gm * N + gn] = f2bf(v);
        else          Cf[(size_t)gm * N + gn] = v;
      }
}

// -------------------------------------------------- FlashAttention (causal)
// grid: (SEQ/64, BATCH*NHEAD); block 256 = 8 waves
// wave w: rows wm*16 (wm=w>>1), cols wn*32 (wn=w&1); 64 q-rows per block
__global__ __launch_bounds__(256) void flash_attn_kernel(
    const __bf16* __restrict__ Q, const __bf16* __restrict__ Kt,
    const __bf16* __restrict__ V, __bf16* __restrict__ O,
    const float* __restrict__ scale_ptr) {
  __shared__ __bf16 Qs[64 * 72];
  __shared__ __bf16 Ks[64 * 72];
  __shared__ __bf16 Vs[64 * 72];   // transposed: Vs[d][k]
  __shared__ __bf16 Ps[64 * 72];
  __shared__ float  Ss[64 * 65];
  __shared__ float  rowf[64];

  const int tid  = threadIdx.x;
  const int lane = tid & 31, hf = lane >> 4, l16 = lane & 15;
  const int w = tid >> 5, wm = w >> 1, wn = w & 1;
  const int q0 = blockIdx.x * 64;
  const int bh = blockIdx.y;
  const int b = bh >> 4, h = bh & 15;
  const float scale = scale_ptr[0];
  const size_t rowbase = (size_t)b * SEQ;

  const int r4 = tid >> 2, seg4 = tid & 3;
  const unsigned lq = lds_off(Qs + r4 * 72 + seg4 * 16);
  const unsigned lk = lds_off(Ks + r4 * 72 + seg4 * 16);

  // async Q tile 64x64 (each thread: 16 elements = 32B)
  async_copy32(lq, Q + (rowbase + q0 + r4) * D_MODEL + h * HD + seg4 * 16);

  v8f acc[2];
  acc[0] = {}; acc[1] = {};
  float m_i = NEG_BIG, l_i = 0.0f;

  const int nkb = blockIdx.x + 1;  // causal: key blocks 0..q0/64
  for (int kb = 0; kb < nkb; ++kb) {
    const int kbase = kb * 64;
    // async K tile
    async_copy32(lk, Kt + (rowbase + kbase + r4) * D_MODEL + h * HD + seg4 * 16);
    {  // V tile, transposed into LDS (element stores)
      const __bf16* g = V + (rowbase + kbase + r4) * D_MODEL + h * HD + seg4 * 16;
#pragma unroll
      for (int j = 0; j < 16; ++j) Vs[(seg4 * 16 + j) * 72 + r4] = g[j];
    }
    wait_async0();
    __syncthreads();

    // S = Q * K^T  (contract hd=64 -> 2 WMMA steps)
    v8f sacc[2];
    sacc[0] = {}; sacc[1] = {};
#pragma unroll
    for (int ks = 0; ks < 2; ++ks) {
      const __bf16* ap = Qs + (wm * 16 + l16) * 72 + ks * 32 + hf * 8;
      v16bf afrag = combine16(*(const v8bf*)ap, *(const v8bf*)(ap + 16));
#pragma unroll
      for (int ni = 0; ni < 2; ++ni) {
        const __bf16* bp = Ks + (wn * 32 + ni * 16 + l16) * 72 + ks * 32 + hf * 8;
        v16bf bfrag = combine16(*(const v8bf*)bp, *(const v8bf*)(bp + 16));
        sacc[ni] = __builtin_amdgcn_wmma_f32_16x16x32_bf16(
            false, afrag, false, bfrag, (short)0, sacc[ni], false, false);
      }
    }
    // scale + causal mask -> Ss
#pragma unroll
    for (int ni = 0; ni < 2; ++ni)
#pragma unroll
      for (int r = 0; r < 8; ++r) {
        int rloc = wm * 16 + r + hf * 8;
        int kg = kbase + wn * 32 + ni * 16 + l16;
        float sv = sacc[ni][r] * scale;
        if (kg > q0 + rloc) sv = NEG_BIG;
        Ss[rloc * 65 + wn * 32 + ni * 16 + l16] = sv;
      }
    __syncthreads();

    // online softmax: one thread per row
    if (tid < 64) {
      const float* srow = Ss + tid * 65;
      float mnew = m_i;
#pragma unroll 8
      for (int j = 0; j < 64; ++j) mnew = fmaxf(mnew, srow[j]);
      float alpha = __expf(m_i - mnew);
      float lsum = l_i * alpha;
      __bf16* prow = Ps + tid * 72;
#pragma unroll 8
      for (int j = 0; j < 64; ++j) {
        float p = __expf(srow[j] - mnew);
        lsum += p;
        prow[j] = f2bf(p);
      }
      m_i = mnew; l_i = lsum;
      rowf[tid] = alpha;
    }
    __syncthreads();

    // rescale accumulators by alpha
#pragma unroll
    for (int r = 0; r < 8; ++r) {
      float a = rowf[wm * 16 + r + hf * 8];
      acc[0][r] *= a;
      acc[1][r] *= a;
    }
    // O += P * V  (contract keys=64 -> 2 WMMA steps)
#pragma unroll
    for (int ks = 0; ks < 2; ++ks) {
      const __bf16* ap = Ps + (wm * 16 + l16) * 72 + ks * 32 + hf * 8;
      v16bf afrag = combine16(*(const v8bf*)ap, *(const v8bf*)(ap + 16));
#pragma unroll
      for (int ni = 0; ni < 2; ++ni) {
        const __bf16* bp = Vs + (wn * 32 + ni * 16 + l16) * 72 + ks * 32 + hf * 8;
        v16bf bfrag = combine16(*(const v8bf*)bp, *(const v8bf*)(bp + 16));
        acc[ni] = __builtin_amdgcn_wmma_f32_16x16x32_bf16(
            false, afrag, false, bfrag, (short)0, acc[ni], false, false);
      }
    }
    __syncthreads();
  }

  if (tid < 64) rowf[tid] = 1.0f / l_i;
  __syncthreads();
#pragma unroll
  for (int ni = 0; ni < 2; ++ni)
#pragma unroll
    for (int r = 0; r < 8; ++r) {
      int rloc = wm * 16 + r + hf * 8;
      float v = acc[ni][r] * rowf[rloc];
      O[(rowbase + q0 + rloc) * D_MODEL + h * HD + wn * 32 + ni * 16 + l16] = f2bf(v);
    }
}

// ------------------------------------------------------------------- launch
extern "C" void kernel_launch(void* const* d_in, const int* in_sizes, int n_in,
                              void* d_out, int out_size, void* d_ws, size_t ws_size,
                              hipStream_t stream) {
  const float* hs    = (const float*)d_in[0];
  const float* wq    = (const float*)d_in[1];
  const float* wk    = (const float*)d_in[2];
  const float* wv    = (const float*)d_in[3];
  const float* wo    = (const float*)d_in[4];
  const float* scale = (const float*)d_in[5];
  float* out = (float*)d_out;

  char* ws = (char*)d_ws;
  const size_t MB = 1024 * 1024;
  __bf16* Xb  = (__bf16*)(ws);             // 8 MB  hidden bf16
  __bf16* Wqb = (__bf16*)(ws +  8 * MB);   // 2 MB
  __bf16* Wkb = (__bf16*)(ws + 10 * MB);   // 2 MB
  __bf16* Wvb = (__bf16*)(ws + 12 * MB);   // 2 MB
  __bf16* Wob = (__bf16*)(ws + 14 * MB);   // 2 MB
  __bf16* Qb  = (__bf16*)(ws + 16 * MB);   // 8 MB
  __bf16* Kb  = (__bf16*)(ws + 24 * MB);   // 8 MB
  __bf16* Vb  = (__bf16*)(ws + 32 * MB);   // 8 MB
  __bf16* Ab  = (__bf16*)(ws + 40 * MB);   // 8 MB attn output

  const int M = BATCH * SEQ, N = D_MODEL, K = D_MODEL;

  f32_to_bf16_kernel<<<(M * K + 255) / 256, 256, 0, stream>>>(hs, Xb, M * K);
  f32_to_bf16_kernel<<<(N * K + 255) / 256, 256, 0, stream>>>(wq, Wqb, N * K);
  f32_to_bf16_kernel<<<(N * K + 255) / 256, 256, 0, stream>>>(wk, Wkb, N * K);
  f32_to_bf16_kernel<<<(N * K + 255) / 256, 256, 0, stream>>>(wv, Wvb, N * K);
  f32_to_bf16_kernel<<<(N * K + 255) / 256, 256, 0, stream>>>(wo, Wob, N * K);

  dim3 gg(N / 128, M / 128);
  gemm_bf16_wmma<true><<<gg, 256, 0, stream>>>(Xb, Wqb, nullptr, Qb, M, N, K);
  gemm_bf16_wmma<true><<<gg, 256, 0, stream>>>(Xb, Wkb, nullptr, Kb, M, N, K);
  gemm_bf16_wmma<true><<<gg, 256, 0, stream>>>(Xb, Wvb, nullptr, Vb, M, N, K);

  int nvec = M * NHEAD;
  headnorm_kernel<<<nvec / 256, 256, 0, stream>>>(Qb, nvec);
  headnorm_kernel<<<nvec / 256, 256, 0, stream>>>(Kb, nvec);

  flash_attn_kernel<<<dim3(SEQ / 64, BATCH * NHEAD), 256, 0, stream>>>(Qb, Kb, Vb, Ab, scale);

  gemm_bf16_wmma<false><<<gg, 256, 0, stream>>>(Ab, Wob, out, nullptr, M, N, K);
}
